// HANLayer_21646635172744
// MI455X (gfx1250) — matrix operand order repaired and checked by reading, not compiled, not run on hardware
//
#include <hip/hip_runtime.h>
#include <hip/hip_bf16.h>

// ---------------------------------------------------------------------------
// HAN layer for MI455X (gfx1250, wave32). Heavy GEMMs use v_wmma_f32_16x16x32_bf16
// (bf16 A/B, f32 accum), 1x4 register blocking, and explicit software-pipelined
// (double-buffered) fragment loads so the inner loop never waits to loadcnt 0.
// Semantic-attention GEMM #4 is algebraically removed:
// mean_n(tanh(h@Wp1+b)@Wp2) == (mean_n tanh(h@Wp1+b)) @ Wp2.
// ---------------------------------------------------------------------------

typedef __attribute__((ext_vector_type(16))) __bf16 v16bf;
typedef __attribute__((ext_vector_type(8)))  __bf16 v8bf;
typedef __attribute__((ext_vector_type(8)))  float  v8f;

#define N_NODES 10000
#define NP      10016          // N padded to multiple of 16
#define E_EDGES 160000
#define ET      170000         // E + N self loops
#define IN_C    512
#define OUT_C   1024
#define OUT2    2048
#define NEG_SLOPE 0.2f

// ---------------- small prep kernels ----------------

__global__ void k_cvt_x(const float* __restrict__ x, __bf16* __restrict__ xb) {
  int idx = blockIdx.x * 256 + threadIdx.x;          // over NP*IN_C
  if (idx >= NP * IN_C) return;
  int row = idx / IN_C;
  float v = (row < N_NODES) ? x[idx] : 0.0f;
  xb[idx] = (__bf16)v;
}

// Wt[n][k] = (n<1024 ? W1[k][n] : W2[k][n-1024]); fused [512,2048] weight, transposed
__global__ void k_build_wt(const float* __restrict__ W1, const float* __restrict__ W2,
                           __bf16* __restrict__ Wt) {
  int idx = blockIdx.x * 256 + threadIdx.x;          // over OUT2*IN_C
  if (idx >= OUT2 * IN_C) return;
  int n = idx / IN_C, k = idx % IN_C;
  float v = (n < OUT_C) ? W1[k * OUT_C + n] : W2[k * OUT_C + (n - OUT_C)];
  Wt[idx] = (__bf16)v;
}

// Wp1t[n][k] = Wp1[k][n], 1024x1024
__global__ void k_build_wp1t(const float* __restrict__ Wp1, __bf16* __restrict__ Wp1t) {
  int idx = blockIdx.x * 256 + threadIdx.x;          // over OUT_C*OUT_C
  if (idx >= OUT_C * OUT_C) return;
  int n = idx / OUT_C, k = idx % OUT_C;
  Wp1t[idx] = (__bf16)Wp1[k * OUT_C + n];
}

// ---------------- fragment helpers ----------------

// A 16x32 bf16 frag for lane (mn,hf): VGPR0-3 -> K = hf*8+0..7 ; VGPR4-7 -> K = 16+hf*8+0..7
__device__ __forceinline__ v16bf load_a_frag(const __bf16* arow, int k0, int hf) {
  v8bf lo = *(const v8bf*)(arow + k0 + hf * 8);
  v8bf hi = *(const v8bf*)(arow + k0 + 16 + hf * 8);
  return __builtin_shufflevector(lo, hi, 0,1,2,3,4,5,6,7,8,9,10,11,12,13,14,15);
}
// B 32x16 bf16 frag: lane = col, VGPR0-3 -> K = hf*8+0..7 ; VGPR4-7 -> K = 16+hf*8+0..7
__device__ __forceinline__ v16bf load_b_frag(const __bf16* brow, int k0, int hf) {
  v8bf lo = *(const v8bf*)(brow + k0 + hf * 8);
  v8bf hi = *(const v8bf*)(brow + k0 + 16 + hf * 8);
  return __builtin_shufflevector(lo, hi, 0,1,2,3,4,5,6,7,8,9,10,11,12,13,14,15);
}

// Software-pipelined 1x4-blocked WMMA K-loop shared by both GEMMs.
// Loads for step k+32 are issued BEFORE the 4 wmmas consuming step k, so the
// s_wait before each wmma keeps >=10 loads in flight (no loadcnt-0 stalls).
__device__ __forceinline__ void gemm_kloop(const __bf16* __restrict__ arow,
                                           const __bf16* const brow[4],
                                           int K, int hf, v8f acc[4]) {
  v16bf af = load_a_frag(arow, 0, hf);
  v16bf bf0 = load_b_frag(brow[0], 0, hf);
  v16bf bf1 = load_b_frag(brow[1], 0, hf);
  v16bf bf2 = load_b_frag(brow[2], 0, hf);
  v16bf bf3 = load_b_frag(brow[3], 0, hf);
  for (int k0 = 0; k0 < K - 32; k0 += 32) {
    // stage next k-step's fragments first (double buffer)
    v16bf af_n  = load_a_frag(arow, k0 + 32, hf);
    v16bf bf0_n = load_b_frag(brow[0], k0 + 32, hf);
    v16bf bf1_n = load_b_frag(brow[1], k0 + 32, hf);
    v16bf bf2_n = load_b_frag(brow[2], k0 + 32, hf);
    v16bf bf3_n = load_b_frag(brow[3], k0 + 32, hf);
    __builtin_prefetch(arow + k0 + 64, 0, 3);
    acc[0] = __builtin_amdgcn_wmma_f32_16x16x32_bf16(false, af, false, bf0,
                                                     (short)0, acc[0], false, false);
    acc[1] = __builtin_amdgcn_wmma_f32_16x16x32_bf16(false, af, false, bf1,
                                                     (short)0, acc[1], false, false);
    acc[2] = __builtin_amdgcn_wmma_f32_16x16x32_bf16(false, af, false, bf2,
                                                     (short)0, acc[2], false, false);
    acc[3] = __builtin_amdgcn_wmma_f32_16x16x32_bf16(false, af, false, bf3,
                                                     (short)0, acc[3], false, false);
    af = af_n; bf0 = bf0_n; bf1 = bf1_n; bf2 = bf2_n; bf3 = bf3_n;
  }
  acc[0] = __builtin_amdgcn_wmma_f32_16x16x32_bf16(false, af, false, bf0,
                                                   (short)0, acc[0], false, false);
  acc[1] = __builtin_amdgcn_wmma_f32_16x16x32_bf16(false, af, false, bf1,
                                                   (short)0, acc[1], false, false);
  acc[2] = __builtin_amdgcn_wmma_f32_16x16x32_bf16(false, af, false, bf2,
                                                   (short)0, acc[2], false, false);
  acc[3] = __builtin_amdgcn_wmma_f32_16x16x32_bf16(false, af, false, bf3,
                                                   (short)0, acc[3], false, false);
}

// ---------------- WMMA GEMM: H[NP,2048] = xb[NP,512] @ [W1|W2] ----------------
// Each wave owns a 16x64 strip of C.

__global__ void k_gemm_x(const __bf16* __restrict__ A,   // [NP,512]
                         const __bf16* __restrict__ Bt,  // [2048,512] (transposed)
                         float* __restrict__ C) {        // [NP,2048]
  const int K = IN_C;
  const int ngroups = OUT2 / 64;                         // 32 groups of 4 N-tiles
  int wave = blockIdx.x * (blockDim.x >> 5) + (threadIdx.x >> 5);
  int tm = wave / ngroups, tg = wave % ngroups;
  if (tm >= NP / 16) return;
  int lane = threadIdx.x & 31;
  int hf   = lane >> 4;                                  // half-wave select
  int mn   = lane & 15;                                  // row of A / col of B

  const __bf16* arow = A + (size_t)(tm * 16 + mn) * K;
  const __bf16* brow[4];
#pragma unroll
  for (int j = 0; j < 4; ++j)
    brow[j] = Bt + (size_t)((tg * 4 + j) * 16 + mn) * K;

  v8f acc[4] = {};
  gemm_kloop(arow, brow, K, hf, acc);

  // D layout: VGPR r -> row = tm*16 + hf*8 + r, col = tile*16 + (lane&15)
#pragma unroll
  for (int j = 0; j < 4; ++j) {
    float* crow = C + (size_t)(tm * 16 + hf * 8) * OUT2 + (tg * 4 + j) * 16 + mn;
#pragma unroll
    for (int r = 0; r < 8; ++r) crow[(size_t)r * OUT2] = acc[j][r];
  }
}

// ---------------- attention coefficient matvecs ----------------

__global__ void k_alpha(const float* __restrict__ H,
                        const float* __restrict__ as1, const float* __restrict__ ad1,
                        const float* __restrict__ as2, const float* __restrict__ ad2,
                        float* A_s1, float* A_d1, float* A_s2, float* A_d2) {
  int node = blockIdx.x;                 // < N_NODES
  int w    = threadIdx.x >> 5;           // 0..3 : which dot product
  int lane = threadIdx.x & 31;
  const float* h = H + (size_t)node * OUT2 + ((w >= 2) ? OUT_C : 0);
  const float* a = (w == 0) ? as1 : (w == 1) ? ad1 : (w == 2) ? as2 : ad2;
  float s = 0.0f;
  for (int c = lane; c < OUT_C; c += 32) s += h[c] * a[c];
  for (int off = 16; off; off >>= 1) s += __shfl_xor(s, off, 32);
  if (lane == 0) {
    float* dst = (w == 0) ? A_s1 : (w == 1) ? A_d1 : (w == 2) ? A_s2 : A_d2;
    dst[node] = s;
  }
}

// ---------------- edge softmax ----------------

__device__ __forceinline__ unsigned f2o(float f) {       // order-preserving f32->u32
  unsigned u = __float_as_uint(f);
  return (u & 0x80000000u) ? ~u : (u | 0x80000000u);
}
__device__ __forceinline__ float o2f(unsigned u) {
  return (u & 0x80000000u) ? __uint_as_float(u ^ 0x80000000u) : __uint_as_float(~u);
}
__device__ __forceinline__ float leaky(float e) {
  return (e >= 0.0f) ? e : NEG_SLOPE * e;
}

__global__ void k_edge_max(const int* __restrict__ ei,
                           const float* As1, const float* Ad1,
                           const float* As2, const float* Ad2,
                           unsigned* __restrict__ m1, unsigned* __restrict__ m2) {
  int e = blockIdx.x * 256 + threadIdx.x;
  if (e >= ET) return;
  int src, dst;
  if (e < E_EDGES) { src = ei[e]; dst = ei[E_EDGES + e]; } else { src = dst = e - E_EDGES; }
  atomicMax(&m1[dst], f2o(leaky(As1[src] + Ad1[dst])));
  atomicMax(&m2[dst], f2o(leaky(As2[src] + Ad2[dst])));
}

__global__ void k_edge_sum(const int* __restrict__ ei,
                           const float* As1, const float* Ad1,
                           const float* As2, const float* Ad2,
                           const unsigned* __restrict__ m1, const unsigned* __restrict__ m2,
                           float* __restrict__ ex1, float* __restrict__ ex2,
                           float* __restrict__ den1, float* __restrict__ den2) {
  int e = blockIdx.x * 256 + threadIdx.x;
  if (e >= ET) return;
  int src, dst;
  if (e < E_EDGES) { src = ei[e]; dst = ei[E_EDGES + e]; } else { src = dst = e - E_EDGES; }
  float v1 = __expf(leaky(As1[src] + Ad1[dst]) - o2f(m1[dst]));
  float v2 = __expf(leaky(As2[src] + Ad2[dst]) - o2f(m2[dst]));
  ex1[e] = v1; ex2[e] = v2;
  atomicAdd(&den1[dst], v1);
  atomicAdd(&den2[dst], v2);
}

// one block per edge; H rows live in the 192MB L2 so gathers stay on-chip
__global__ void k_aggregate(const int* __restrict__ ei,
                            const float* __restrict__ ex1, const float* __restrict__ ex2,
                            const float* __restrict__ den1, const float* __restrict__ den2,
                            const float* __restrict__ H,
                            float* __restrict__ O1, float* __restrict__ O2) {
  int e = blockIdx.x;
  int src, dst;
  if (e < E_EDGES) { src = ei[e]; dst = ei[E_EDGES + e]; } else { src = dst = e - E_EDGES; }
  float a1 = ex1[e] / den1[dst];
  float a2 = ex2[e] / den2[dst];
  const float* hs = H + (size_t)src * OUT2;
  float* o1 = O1 + (size_t)dst * OUT_C;
  float* o2 = O2 + (size_t)dst * OUT_C;
  for (int c = threadIdx.x; c < OUT_C; c += blockDim.x) {
    atomicAdd(&o1[c], a1 * hs[c]);
    atomicAdd(&o2[c], a2 * hs[OUT_C + c]);
  }
}

// ---------------- PReLU + fuse, bf16 copy for semantic GEMM ----------------

__global__ void k_combine(float* __restrict__ O1, float* __restrict__ O2,
                          const float* __restrict__ b1, const float* __restrict__ b2,
                          const float* __restrict__ prelu_a,
                          __bf16* __restrict__ hsum) {
  int idx = blockIdx.x * 256 + threadIdx.x;        // over NP*OUT_C
  if (idx >= NP * OUT_C) return;
  int row = idx / OUT_C, col = idx % OUT_C;
  if (row >= N_NODES) { O1[idx] = 0.0f; O2[idx] = 0.0f; hsum[idx] = (__bf16)0.0f; return; }
  float pa = prelu_a[0];
  float v1 = O1[idx] + b1[col]; v1 = (v1 >= 0.0f) ? v1 : pa * v1;
  float v2 = O2[idx] + b2[col]; v2 = (v2 >= 0.0f) ? v2 : pa * v2;
  O1[idx] = v1; O2[idx] = v2;
  hsum[idx] = (__bf16)(v1 + v2);
}

// ---------------- WMMA GEMM #2 with fused tanh + masked column-sum ----------------
// Output tiles are never materialized, only the masked column sums needed for
// the semantic-attention mean.

__global__ void k_gemm_p(const __bf16* __restrict__ A,   // hsum [NP,1024]
                         const __bf16* __restrict__ Bt,  // Wp1t [1024,1024]
                         const float* __restrict__ bp1,
                         float* __restrict__ colsum) {
  const int K = OUT_C;
  const int ngroups = OUT_C / 64;                        // 16 groups of 4 N-tiles
  int wave = blockIdx.x * (blockDim.x >> 5) + (threadIdx.x >> 5);
  int tm = wave / ngroups, tg = wave % ngroups;
  if (tm >= NP / 16) return;
  int lane = threadIdx.x & 31;
  int hf   = lane >> 4;
  int mn   = lane & 15;

  const __bf16* arow = A + (size_t)(tm * 16 + mn) * K;
  const __bf16* brow[4];
#pragma unroll
  for (int j = 0; j < 4; ++j)
    brow[j] = Bt + (size_t)((tg * 4 + j) * 16 + mn) * K;

  v8f acc[4] = {};
  gemm_kloop(arow, brow, K, hf, acc);

  int rbase = tm * 16 + hf * 8;
#pragma unroll
  for (int j = 0; j < 4; ++j) {
    int col = (tg * 4 + j) * 16 + mn;
    float bias = bp1[col];
    float partial = 0.0f;
#pragma unroll
    for (int r = 0; r < 8; ++r) {
      float t = tanhf(acc[j][r] + bias);
      if (rbase + r < N_NODES) partial += t;     // exclude padded rows from the mean
    }
    partial += __shfl_xor(partial, 16, 32);      // combine the two half-waves (same col)
    if (hf == 0) atomicAdd(&colsum[col], partial);
  }
}

// ---------------- tiny matvec + softmax over 1024 dims ----------------

__global__ void k_semantic(const float* __restrict__ colsum,
                           const float* __restrict__ Wp2,   // [1024,1024]
                           float* __restrict__ att) {
  __shared__ float red[256];
  __shared__ float tm[OUT_C];
  int t = threadIdx.x;
  for (int k = t; k < OUT_C; k += 256) tm[k] = colsum[k] * (1.0f / (float)N_NODES);
  __syncthreads();
  float w[4];
#pragma unroll
  for (int j = 0; j < 4; ++j) {
    int n = t + j * 256;
    float s = 0.0f;
    for (int k = 0; k < OUT_C; ++k) s += tm[k] * Wp2[(size_t)k * OUT_C + n];
    w[j] = s;
  }
  float m = fmaxf(fmaxf(w[0], w[1]), fmaxf(w[2], w[3]));
  red[t] = m; __syncthreads();
  for (int off = 128; off; off >>= 1) { if (t < off) red[t] = fmaxf(red[t], red[t + off]); __syncthreads(); }
  m = red[0]; __syncthreads();
  float e[4], s = 0.0f;
#pragma unroll
  for (int j = 0; j < 4; ++j) { e[j] = __expf(w[j] - m); s += e[j]; }
  red[t] = s; __syncthreads();
  for (int off = 128; off; off >>= 1) { if (t < off) red[t] += red[t + off]; __syncthreads(); }
  s = red[0];
#pragma unroll
  for (int j = 0; j < 4; ++j) att[t + j * 256] = e[j] / s;
}

__global__ void k_output(const float* __restrict__ h1, const float* __restrict__ h2,
                         const float* __restrict__ att, float* __restrict__ out) {
  int idx = blockIdx.x * 256 + threadIdx.x;       // over N_NODES*OUT_C
  if (idx >= N_NODES * OUT_C) return;
  int col = idx % OUT_C;
  float a = att[col];
  out[idx] = a * h1[idx] + (1.0f - a) * h2[idx];
}

// ---------------------------------------------------------------------------

extern "C" void kernel_launch(void* const* d_in, const int* in_sizes, int n_in,
                              void* d_out, int out_size, void* d_ws, size_t ws_size,
                              hipStream_t stream) {
  const float* x     = (const float*)d_in[0];
  const int*   ei    = (const int*)  d_in[1];
  const float* W1    = (const float*)d_in[2];
  const float* as1   = (const float*)d_in[3];
  const float* ad1   = (const float*)d_in[4];
  const float* b1    = (const float*)d_in[5];
  const float* W2    = (const float*)d_in[6];
  const float* as2   = (const float*)d_in[7];
  const float* ad2   = (const float*)d_in[8];
  const float* b2    = (const float*)d_in[9];
  const float* prelu = (const float*)d_in[10];
  const float* Wp1   = (const float*)d_in[11];
  const float* bp1   = (const float*)d_in[12];
  const float* Wp2   = (const float*)d_in[13];
  float* out = (float*)d_out;

  // ---- carve workspace ----
  char* p = (char*)d_ws;
  auto carve = [&](size_t bytes) { void* r = (void*)p; p += (bytes + 255) & ~(size_t)255; return r; };
  __bf16*  xb     = (__bf16*) carve((size_t)NP * IN_C * 2);
  __bf16*  Wt     = (__bf16*) carve((size_t)OUT2 * IN_C * 2);
  __bf16*  Wp1t   = (__bf16*) carve((size_t)OUT_C * OUT_C * 2);
  float*   H      = (float*)  carve((size_t)NP * OUT2 * 4);
  float*   O1     = (float*)  carve((size_t)NP * OUT_C * 4);
  float*   O2     = (float*)  carve((size_t)NP * OUT_C * 4);
  __bf16*  hsum   = (__bf16*) carve((size_t)NP * OUT_C * 2);
  float*   A_s1   = (float*)  carve((size_t)N_NODES * 4);
  float*   A_d1   = (float*)  carve((size_t)N_NODES * 4);
  float*   A_s2   = (float*)  carve((size_t)N_NODES * 4);
  float*   A_d2   = (float*)  carve((size_t)N_NODES * 4);
  unsigned* m1    = (unsigned*)carve((size_t)N_NODES * 4);
  unsigned* m2    = (unsigned*)carve((size_t)N_NODES * 4);
  float*   den1   = (float*)  carve((size_t)N_NODES * 4);
  float*   den2   = (float*)  carve((size_t)N_NODES * 4);
  float*   ex1    = (float*)  carve((size_t)ET * 4);
  float*   ex2    = (float*)  carve((size_t)ET * 4);
  float*   colsum = (float*)  carve((size_t)OUT_C * 4);
  float*   att    = (float*)  carve((size_t)OUT_C * 4);

  // ---- zero-init (graph-capturable memsets) ----
  hipMemsetAsync(m1,   0, (size_t)N_NODES * 4, stream);   // encoded 0 < any real encoding
  hipMemsetAsync(m2,   0, (size_t)N_NODES * 4, stream);
  hipMemsetAsync(den1, 0, (size_t)N_NODES * 4, stream);
  hipMemsetAsync(den2, 0, (size_t)N_NODES * 4, stream);
  hipMemsetAsync(O1,   0, (size_t)NP * OUT_C * 4, stream);
  hipMemsetAsync(O2,   0, (size_t)NP * OUT_C * 4, stream);
  hipMemsetAsync(colsum, 0, (size_t)OUT_C * 4, stream);

  // ---- prep: bf16 conversions / transposes ----
  k_cvt_x   <<<(NP * IN_C + 255) / 256, 256, 0, stream>>>(x, xb);
  k_build_wt<<<(OUT2 * IN_C + 255) / 256, 256, 0, stream>>>(W1, W2, Wt);
  k_build_wp1t<<<(OUT_C * OUT_C + 255) / 256, 256, 0, stream>>>(Wp1, Wp1t);

  // ---- fused GEMM: H = x @ [W1|W2]  (bf16 WMMA, f32 accum, 16x64 per wave) ----
  {
    int waves = (NP / 16) * (OUT2 / 64);                  // 626*32
    k_gemm_x<<<(waves + 7) / 8, 256, 0, stream>>>(xb, Wt, H);
  }

  // ---- per-node attention coefficients ----
  k_alpha<<<N_NODES, 128, 0, stream>>>(H, as1, ad1, as2, ad2, A_s1, A_d1, A_s2, A_d2);

  // ---- edge softmax + aggregation (H stays resident in L2) ----
  k_edge_max<<<(ET + 255) / 256, 256, 0, stream>>>(ei, A_s1, A_d1, A_s2, A_d2, m1, m2);
  k_edge_sum<<<(ET + 255) / 256, 256, 0, stream>>>(ei, A_s1, A_d1, A_s2, A_d2,
                                                   m1, m2, ex1, ex2, den1, den2);
  k_aggregate<<<ET, 256, 0, stream>>>(ei, ex1, ex2, den1, den2, H, O1, O2);

  // ---- PReLU + h = h1p + h2p (bf16 copy for next GEMM) ----
  k_combine<<<(NP * OUT_C + 255) / 256, 256, 0, stream>>>(O1, O2, b1, b2, prelu, hsum);

  // ---- GEMM #2 with fused tanh + masked column mean (GEMM #4 eliminated) ----
  {
    int waves = (NP / 16) * (OUT_C / 64);                 // 626*16
    k_gemm_p<<<(waves + 7) / 8, 256, 0, stream>>>(hsum, Wp1t, bp1, colsum);
  }

  // ---- semantic attention (matvec + softmax) and final blend ----
  k_semantic<<<1, 256, 0, stream>>>(colsum, Wp2, att);
  k_output<<<(N_NODES * OUT_C + 255) / 256, 256, 0, stream>>>(O1, O2, att, out);
}
